// BeamSearchDecoder_16836271800404
// MI455X (gfx1250) — matrix operand order repaired and checked by reading, not compiled, not run on hardware
//
#include <hip/hip_runtime.h>
#include <hip/hip_bf16.h>
#include <math.h>

// ---------------- problem constants ----------------
#define B_      8
#define K_      8
#define BK      64
#define TENC    256
#define H_      512
#define V_      2000
#define MAXLEN  128
#define SOS_T   1
#define EOS_T   2
#define MIN_LEN 5
#define NEGV    (-1e30f)

// ---------------- launch shape ----------------
#define NBLK  32
#define NTHR  256
#define NWAVE (NBLK * (NTHR / 32))   // 256 waves

// ---------------- types ----------------
typedef __bf16 bf16_t;
typedef __attribute__((ext_vector_type(16))) __bf16 bf16x16;
typedef __attribute__((ext_vector_type(8)))  float  f32x8;

typedef union { bf16x16 v; uint4 q[2]; } FragU;

// ---------------- workspace layout (bytes, all 16B aligned) ----------------
constexpr size_t OFF_OUTS   = 0;                                            // [MAXLEN][BK][V] f32
constexpr size_t OFF_EMBB   = OFF_OUTS   + (size_t)MAXLEN * BK * V_ * 4;    // [V][H] bf16
constexpr size_t OFF_WXHT   = OFF_EMBB   + (size_t)V_ * H_ * 2;             // [1536][1024] bf16 (N-major, k<512=Wx,k>=512=Wh)
constexpr size_t OFF_WOT    = OFF_WXHT   + (size_t)1536 * 1024 * 2;         // [2000][1024] bf16 (N-major)
constexpr size_t OFF_ENCB   = OFF_WOT    + (size_t)V_ * 1024 * 2;           // [8][256][512] bf16
constexpr size_t OFF_ENCT   = OFF_ENCB   + (size_t)B_ * TENC * H_ * 2;      // [8][512][256] bf16 (transposed)
constexpr size_t OFF_GXH    = OFF_ENCT   + (size_t)B_ * H_ * TENC * 2;      // [64][3072] f32 (gx | gh)
constexpr size_t OFF_H      = OFF_GXH    + (size_t)BK * 3072 * 4;           // [64][512] f32
constexpr size_t OFF_H2     = OFF_H      + (size_t)BK * H_ * 4;             // [64][512] f32
constexpr size_t OFF_HB     = OFF_H2     + (size_t)BK * H_ * 4;             // [64][512] bf16
constexpr size_t OFF_H2CB   = OFF_HB     + (size_t)BK * H_ * 2;             // [64][1024] bf16  (h2 | ctx)
constexpr size_t OFF_SCORES = OFF_H2CB   + (size_t)BK * 1024 * 2;           // [64][256] f32
constexpr size_t OFF_ATTB   = OFF_SCORES + (size_t)BK * TENC * 4;           // [64][256] bf16
constexpr size_t OFF_LOGITS = OFF_ATTB   + (size_t)BK * TENC * 2;           // [64][2000] f32
constexpr size_t OFF_TOK    = OFF_LOGITS + (size_t)BK * V_ * 4;             // [64] i32 (padded)
constexpr size_t OFF_PREDS  = OFF_TOK    + 256;                             // [MAXLEN][64] i32
constexpr size_t OFF_PENALS = OFF_PREDS  + (size_t)MAXLEN * BK * 4;         // [MAXLEN][64] f32
constexpr size_t OFF_SEQSC  = OFF_PENALS + (size_t)MAXLEN * BK * 4;         // [64] f32 (padded)
constexpr size_t OFF_SEL    = OFF_SEQSC  + 256;                             // [MAXLEN][8] i32
constexpr size_t OFF_BAR    = OFF_SEL    + (size_t)MAXLEN * B_ * 4;         // 2x u32 barrier

// ---------------- WMMA helpers ----------------
__device__ __forceinline__ f32x8 wmma_bf16(bf16x16 a, bf16x16 b, f32x8 c) {
  return __builtin_amdgcn_wmma_f32_16x16x32_bf16(false, a, false, b, (short)0, c, false, false);
}

// A fragment (16x32 bf16): lane L (l=L&15, hi=L>>4) holds row m0+l.
// lanes 0-15: K = k0..k0+15 ; lanes 16-31: K = k0+16..k0+31 (contiguous per lane)
__device__ __forceinline__ bf16x16 load_a(const bf16_t* row_ptr, int k0, int hi) {
  FragU f;
  const bf16_t* p = row_ptr + k0 + hi * 16;
  f.q[0] = *(const uint4*)(p);
  f.q[1] = *(const uint4*)(p + 8);
  return f.v;
}

// B fragment (32x16 bf16) from N-major matrix: lane holds column n0+l.
// lanes 0-15: K = k0..k0+15 ; lanes 16-31: K = k0+16..k0+31 (contiguous per lane)
__device__ __forceinline__ bf16x16 load_b(const bf16_t* col_ptr, int k0, int hi) {
  FragU f;
  const bf16_t* p = col_ptr + k0 + hi * 16;
  f.q[0] = *(const uint4*)(p);
  f.q[1] = *(const uint4*)(p + 8);
  return f.v;
}

__device__ __forceinline__ float sigmoidf_(float x) { return 1.0f / (1.0f + expf(-x)); }

// ---------------- device-wide barrier (sense reversal via generation) ----------------
__device__ __forceinline__ void gbar(unsigned* bar) {
  __threadfence();
  __syncthreads();
  if (threadIdx.x == 0) {
    unsigned* cnt = bar;
    unsigned* gen = bar + 1;
    unsigned g = __hip_atomic_load(gen, __ATOMIC_RELAXED, __HIP_MEMORY_SCOPE_AGENT);
    unsigned a = __hip_atomic_fetch_add(cnt, 1u, __ATOMIC_ACQ_REL, __HIP_MEMORY_SCOPE_AGENT);
    if (a == NBLK - 1) {
      __hip_atomic_store(cnt, 0u, __ATOMIC_RELAXED, __HIP_MEMORY_SCOPE_AGENT);
      __hip_atomic_fetch_add(gen, 1u, __ATOMIC_ACQ_REL, __HIP_MEMORY_SCOPE_AGENT);
    } else {
      while (__hip_atomic_load(gen, __ATOMIC_ACQUIRE, __HIP_MEMORY_SCOPE_AGENT) == g) {
        __builtin_amdgcn_s_sleep(1);
      }
    }
  }
  __syncthreads();
  __threadfence();
}

// ============================================================================
__global__ __launch_bounds__(NTHR, 1)
void beam_decode_kernel(const float* __restrict__ enc, const float* __restrict__ emb,
                        const float* __restrict__ Wx,  const float* __restrict__ Wh,
                        const float* __restrict__ bvec,const float* __restrict__ Wo,
                        const float* __restrict__ bo,  float* __restrict__ out,
                        char* __restrict__ ws)
{
  float*  outs   = (float*) (ws + OFF_OUTS);
  bf16_t* embB   = (bf16_t*)(ws + OFF_EMBB);
  bf16_t* wxhT   = (bf16_t*)(ws + OFF_WXHT);
  bf16_t* woT    = (bf16_t*)(ws + OFF_WOT);
  bf16_t* encB   = (bf16_t*)(ws + OFF_ENCB);
  bf16_t* encT   = (bf16_t*)(ws + OFF_ENCT);
  float*  gxh    = (float*) (ws + OFF_GXH);
  float*  hbuf   = (float*) (ws + OFF_H);
  float*  h2buf  = (float*) (ws + OFF_H2);
  bf16_t* hB     = (bf16_t*)(ws + OFF_HB);
  bf16_t* h2cB   = (bf16_t*)(ws + OFF_H2CB);
  float*  scores = (float*) (ws + OFF_SCORES);
  bf16_t* attB   = (bf16_t*)(ws + OFF_ATTB);
  float*  logits = (float*) (ws + OFF_LOGITS);
  int*    tok    = (int*)   (ws + OFF_TOK);
  int*    preds  = (int*)   (ws + OFF_PREDS);
  float*  penals = (float*) (ws + OFF_PENALS);
  float*  seqsc  = (float*) (ws + OFF_SEQSC);
  int*    sel    = (int*)   (ws + OFF_SEL);
  unsigned* bar  = (unsigned*)(ws + OFF_BAR);

  __shared__ float red_v[NTHR];
  __shared__ int   red_i[NTHR];
  __shared__ int   chosen[K_];
  __shared__ float chosen_v[K_];

  const int tid  = blockIdx.x * NTHR + threadIdx.x;
  const int nth  = NBLK * NTHR;
  const int wave = tid >> 5;
  const int lane = threadIdx.x & 31;
  const int l    = lane & 15;
  const int hi   = lane >> 4;

  // ---------------- Phase 0: bf16 conversions / transposes / state init ----------
  for (int i = tid; i < V_ * H_; i += nth) embB[i] = (bf16_t)emb[i];
  for (int i = tid; i < 1536 * 1024; i += nth) {
    int n = i >> 10, k = i & 1023;
    float v = (k < H_) ? Wx[k * 1536 + n] : Wh[(k - H_) * 1536 + n];
    wxhT[i] = (bf16_t)v;
  }
  for (int i = tid; i < V_ * 1024; i += nth) {
    int n = i >> 10, k = i & 1023;
    woT[i] = (bf16_t)Wo[(size_t)k * V_ + n];
  }
  for (int i = tid; i < B_ * TENC * H_; i += nth) encB[i] = (bf16_t)enc[i];
  for (int i = tid; i < B_ * H_ * TENC; i += nth) {
    int b  = i / (H_ * TENC);
    int r  = i % (H_ * TENC);
    int hd = r / TENC, tt = r % TENC;
    encT[i] = (bf16_t)enc[((size_t)b * TENC + tt) * H_ + hd];
  }
  for (int i = tid; i < BK; i += nth) {
    tok[i]   = SOS_T;
    seqsc[i] = ((i & 7) == 0) ? 0.0f : NEGV;
  }
  for (int i = tid; i < BK * H_; i += nth) { hbuf[i] = 0.0f; hB[i] = (bf16_t)0.0f; }
  gbar(bar);

  // ============================ main decode loop ============================
  for (int t = 0; t < MAXLEN; ++t) {
    // -------- P1: gxh = [ x@Wx | h@Wh ]  (64 x 3072), K=512 each --------
    for (int tile = wave; tile < 4 * 192; tile += NWAVE) {
      int mt = tile / 192, nt = tile % 192;
      bool isGh = (nt >= 96);
      int  cb   = (isGh ? (nt - 96) : nt) * 16;
      int  m    = mt * 16 + l;
      const bf16_t* brow = wxhT + (size_t)(cb + l) * 1024 + (isGh ? H_ : 0);
      const bf16_t* arow = isGh ? (hB + (size_t)m * H_) : (embB + (size_t)tok[m] * H_);
      f32x8 c = {};
#pragma unroll 4
      for (int ki = 0; ki < 16; ++ki)
        c = wmma_bf16(load_a(arow, ki * 32, hi), load_b(brow, ki * 32, hi), c);
      int obase = isGh ? (1536 + cb) : cb;
#pragma unroll
      for (int r = 0; r < 8; ++r)
        gxh[(size_t)(mt * 16 + r + hi * 8) * 3072 + obase + l] = c[r];
    }
    gbar(bar);

    // -------- P2: GRU elementwise -> h2, h2 half of h2cB --------
    for (int i = tid; i < BK * H_; i += nth) {
      int row = i >> 9, j = i & 511;
      const float* g = gxh + (size_t)row * 3072;
      float z  = sigmoidf_(g[j]        + bvec[j]        + g[1536 + j]);
      float r  = sigmoidf_(g[512 + j]  + bvec[512 + j]  + g[2048 + j]);
      float nn = tanhf   (g[1024 + j] + bvec[1024 + j] + r * g[2560 + j]);
      float h2 = (1.0f - z) * nn + z * hbuf[i];
      h2buf[i] = h2;
      h2cB[(size_t)row * 1024 + j] = (bf16_t)h2;
    }
    gbar(bar);

    // -------- P3: attention scores: per enc-group g (enc[bk]=enc[bk%8]) --------
    for (int tile = wave; tile < 8 * 16; tile += NWAVE) {
      int g = tile >> 4, nt = tile & 15;
      const bf16_t* arow = h2cB + (size_t)(g + 8 * (l & 7)) * 1024;           // h2 part (k<512)
      const bf16_t* brow = encB + ((size_t)g * TENC + nt * 16 + l) * H_;
      f32x8 c = {};
#pragma unroll 4
      for (int ki = 0; ki < 16; ++ki)
        c = wmma_bf16(load_a(arow, ki * 32, hi), load_b(brow, ki * 32, hi), c);
      if (hi == 0) {
#pragma unroll
        for (int r = 0; r < 8; ++r)
          scores[(size_t)(g + 8 * r) * TENC + nt * 16 + l] = c[r];
      }
    }
    gbar(bar);

    // -------- P4: softmax over T=256 (one wave per beam row) --------
    if (wave < BK) {
      const float* s = scores + (size_t)wave * TENC;
      float m = -3.4e38f;
#pragma unroll
      for (int ci = 0; ci < 8; ++ci) m = fmaxf(m, s[lane + ci * 32]);
      for (int o = 16; o > 0; o >>= 1) m = fmaxf(m, __shfl_xor(m, o, 32));
      float e[8]; float sum = 0.0f;
#pragma unroll
      for (int ci = 0; ci < 8; ++ci) { e[ci] = expf(s[lane + ci * 32] - m); sum += e[ci]; }
      for (int o = 16; o > 0; o >>= 1) sum += __shfl_xor(sum, o, 32);
      float inv = 1.0f / sum;
#pragma unroll
      for (int ci = 0; ci < 8; ++ci)
        attB[(size_t)wave * TENC + lane + ci * 32] = (bf16_t)(e[ci] * inv);
    }
    gbar(bar);

    // -------- P5: ctx = att @ enc  -> ctx half of h2cB --------
    for (int tile = wave; tile < 8 * 32; tile += NWAVE) {
      int g = tile >> 5, nt = tile & 31;
      const bf16_t* arow = attB + (size_t)(g + 8 * (l & 7)) * TENC;
      const bf16_t* brow = encT + ((size_t)g * H_ + nt * 16 + l) * TENC;
      f32x8 c = {};
#pragma unroll
      for (int ki = 0; ki < 8; ++ki)
        c = wmma_bf16(load_a(arow, ki * 32, hi), load_b(brow, ki * 32, hi), c);
      if (hi == 0) {
#pragma unroll
        for (int r = 0; r < 8; ++r)
          h2cB[(size_t)(g + 8 * r) * 1024 + 512 + nt * 16 + l] = (bf16_t)c[r];
      }
    }
    gbar(bar);

    // -------- P6: logits = [h2|ctx] @ Wo + bo  (64 x 2000, K=1024) --------
    for (int tile = wave; tile < 4 * 125; tile += NWAVE) {
      int mt = tile / 125, nt = tile % 125;
      const bf16_t* arow = h2cB + (size_t)(mt * 16 + l) * 1024;
      const bf16_t* brow = woT  + (size_t)(nt * 16 + l) * 1024;
      f32x8 c = {};
#pragma unroll 4
      for (int ki = 0; ki < 32; ++ki)
        c = wmma_bf16(load_a(arow, ki * 32, hi), load_b(brow, ki * 32, hi), c);
      int ncol = nt * 16 + l;
      float bias = bo[ncol];
#pragma unroll
      for (int r = 0; r < 8; ++r)
        logits[(size_t)(mt * 16 + r + hi * 8) * V_ + ncol] = c[r] + bias;
    }
    gbar(bar);

    // -------- P7: log_softmax over V + EOS mask, store outs[t] --------
    if (wave < BK) {
      const float* lg = logits + (size_t)wave * V_;
      float m = -3.4e38f;
      for (int c = lane; c < V_; c += 32) m = fmaxf(m, lg[c]);
      for (int o = 16; o > 0; o >>= 1) m = fmaxf(m, __shfl_xor(m, o, 32));
      float sum = 0.0f;
      for (int c = lane; c < V_; c += 32) sum += expf(lg[c] - m);
      for (int o = 16; o > 0; o >>= 1) sum += __shfl_xor(sum, o, 32);
      float lse = m + logf(sum);
      float* orow = outs + ((size_t)t * BK + wave) * V_;
      for (int c = lane; c < V_; c += 32) {
        float v = lg[c] - lse;
        if (t < MIN_LEN && c == EOS_T) v = NEGV;
        orow[c] = v;
      }
    }
    gbar(bar);

    // -------- P8: per-batch top-8 over K*V = 16000 candidates --------
    if (blockIdx.x < B_) {
      int b = blockIdx.x;
      const float* ob = outs + ((size_t)t * BK + b * K_) * V_;   // 8 x 2000 contiguous
      for (int j = 0; j < K_; ++j) {
        float best = -3.4e38f; int bidx = 0x7fffffff;
        for (int c = threadIdx.x; c < K_ * V_; c += NTHR) {
          bool skip = false;
          for (int q = 0; q < j; ++q) skip = skip || (chosen[q] == c);
          if (skip) continue;
          float val = seqsc[b * K_ + (c / V_)] + ob[c];
          if (val > best || (val == best && c < bidx)) { best = val; bidx = c; }
        }
        red_v[threadIdx.x] = best; red_i[threadIdx.x] = bidx;
        __syncthreads();
        for (int s = NTHR / 2; s > 0; s >>= 1) {
          if (threadIdx.x < s) {
            float v2 = red_v[threadIdx.x + s]; int i2 = red_i[threadIdx.x + s];
            if (v2 > red_v[threadIdx.x] ||
                (v2 == red_v[threadIdx.x] && i2 < red_i[threadIdx.x])) {
              red_v[threadIdx.x] = v2; red_i[threadIdx.x] = i2;
            }
          }
          __syncthreads();
        }
        if (threadIdx.x == 0) { chosen[j] = red_i[0]; chosen_v[j] = red_v[0]; }
        __syncthreads();
      }
      if (threadIdx.x == 0) {
        float lp = powf((MIN_LEN + (float)(t + 1)) / (MIN_LEN + 1.0f), 1.2f);
        for (int j = 0; j < K_; ++j) {
          int c  = chosen[j];
          int tk = c % V_;
          int pr = c / V_ + b * K_;
          int bk = b * K_ + j;
          tok[bk]              = tk;
          preds[t * BK + bk]   = pr;
          penals[t * BK + bk]  = chosen_v[j] / lp;
          seqsc[bk]            = (tk == EOS_T) ? NEGV : chosen_v[j];
        }
      }
    }
    gbar(bar);

    // -------- P9: h = h2[pred] --------
    for (int i = tid; i < BK * H_; i += nth) {
      int row = i >> 9, j = i & 511;
      int p = preds[t * BK + row];
      float v = h2buf[(size_t)p * H_ + j];
      hbuf[i] = v;
      hB[i]   = (bf16_t)v;
    }
    gbar(bar);
  }

  // ---------------- backtrack: best beam chain per batch ----------------
  if (tid < B_) {
    int b = tid;
    float best = -3.4e38f; int bi = 0;
    for (int k = 0; k < K_; ++k) {
      float v = penals[(MAXLEN - 1) * BK + b * K_ + k];
      if (v > best) { best = v; bi = k; }
    }
    int tp = b * K_ + bi;
    for (int tt = MAXLEN - 1; tt >= 0; --tt) {
      sel[tt * B_ + b] = tp;
      tp = preds[tt * BK + tp];
    }
  }
  gbar(bar);

  // dec[t][b][:] = outs[t][sel[t][b]][:]
  for (size_t i = tid; i < (size_t)MAXLEN * B_ * V_; i += nth) {
    size_t v  = i % V_;
    size_t tb = i / V_;             // t*8 + b
    size_t tt = tb / B_;
    int s = sel[tb];
    out[i] = outs[(tt * BK + (size_t)s) * V_ + v];
  }
}

// ============================================================================
extern "C" void kernel_launch(void* const* d_in, const int* in_sizes, int n_in,
                              void* d_out, int out_size, void* d_ws, size_t ws_size,
                              hipStream_t stream) {
  (void)in_sizes; (void)n_in; (void)out_size; (void)ws_size;
  // inputs: 0 input_var, 1 encoder_outputs, 2 emb, 3 Wx, 4 Wh, 5 b, 6 Wo, 7 bo
  const float* enc  = (const float*)d_in[1];
  const float* emb  = (const float*)d_in[2];
  const float* Wx   = (const float*)d_in[3];
  const float* Wh   = (const float*)d_in[4];
  const float* bvec = (const float*)d_in[5];
  const float* Wo   = (const float*)d_in[6];
  const float* bo   = (const float*)d_in[7];

  // zero the global-barrier words (graph-capture-safe stream op)
  hipMemsetAsync((char*)d_ws + OFF_BAR, 0, 64, stream);

  beam_decode_kernel<<<NBLK, NTHR, 0, stream>>>(enc, emb, Wx, Wh, bvec, Wo, bo,
                                                (float*)d_out, (char*)d_ws);
}
